// AttnDecoder_22531398435112
// MI455X (gfx1250) — compile-verified, matrix operand order checked by reading, not run on hardware
//
#include <hip/hip_runtime.h>
#include <hip/hip_bf16.h>
#include <math.h>

// ---------------- problem constants (match reference) ----------------
#define BB 64      // batch
#define SS 128     // encoder seq len
#define HH 512     // hidden
#define H2C 1024   // 2H
#define H4C 2048   // 4H
#define VV 32000   // vocab
#define TT 32      // decode steps
#define EE 100     // embedding dim
#define G3 1536    // 3H (gate rows per direction)
#define N3 3072    // 2 * 3H

typedef __bf16 bf16_t;
typedef __attribute__((ext_vector_type(16))) __bf16 v16bf;
typedef __attribute__((ext_vector_type(8)))  __bf16 v8bf;
typedef __attribute__((ext_vector_type(8)))  float  v8f;

union V16U { v16bf v; v8bf h[2]; };

// ---------------- fp32 -> bf16 conversion ----------------
__global__ void k_cvt(const float* __restrict__ src, bf16_t* __restrict__ dst, size_t n) {
    size_t i = (size_t)blockIdx.x * blockDim.x + threadIdx.x;
    if (i < n) dst[i] = (bf16_t)src[i];
}

__global__ void k_copy(const float* __restrict__ src, float* __restrict__ dst, int n) {
    int i = blockIdx.x * blockDim.x + threadIdx.x;
    if (i < n) dst[i] = src[i];
}

// ---------------- bf16 WMMA GEMM: C[M,N] = A[M,K] * Bt[N,K]^T + bias ----------------
// Skinny-M shape: each wave owns one 16-wide N strip and FOUR 16-row M tiles
// (4 accumulators), so the (large) B fragment is loaded once per K-step and
// reused by 4 WMMAs. Fragments are software-pipelined one K-step ahead so
// global loads are covered by the 4 in-flight WMMAs of the previous step.
// Requires M % 64 == 0, N % 64 == 0, K % 32 == 0, K >= 64 (true here).
__global__ __launch_bounds__(128) void wmma_gemm_bf16(
    const bf16_t* __restrict__ A, const bf16_t* __restrict__ Bt,
    const float* __restrict__ bias, float* __restrict__ C,
    int M, int N, int K, long long ldc)
{
    const int wave = threadIdx.x >> 5;
    const int lane = threadIdx.x & 31;
    const int half = lane >> 4;
    const int r    = lane & 15;
    const int nBase = (blockIdx.x * 4 + wave) * 16;
    const int mBase = blockIdx.y * 64;
    if (nBase >= N || mBase >= M) return;

    const bf16_t* bRow = Bt + (size_t)(nBase + r) * K;
    const bf16_t* aRow[4];
#pragma unroll
    for (int m = 0; m < 4; ++m)
        aRow[m] = A + (size_t)(mBase + 16 * m + r) * K;

    const int o0 = half * 8;        // element offset of first 8-wide chunk
    const int o1 = 16 + half * 8;   // element offset of second 8-wide chunk

    v8f acc[4] = {};

    // prologue: fragments for k0 = 0
    V16U bCur, aCur[4];
    bCur.h[0] = *(const v8bf*)(bRow + o0);
    bCur.h[1] = *(const v8bf*)(bRow + o1);
#pragma unroll
    for (int m = 0; m < 4; ++m) {
        aCur[m].h[0] = *(const v8bf*)(aRow[m] + o0);
        aCur[m].h[1] = *(const v8bf*)(aRow[m] + o1);
    }

    for (int k0 = 32; k0 < K; k0 += 32) {
        // issue next-step loads first so they overlap this step's WMMAs
        V16U bNxt, aNxt[4];
        bNxt.h[0] = *(const v8bf*)(bRow + k0 + o0);
        bNxt.h[1] = *(const v8bf*)(bRow + k0 + o1);
#pragma unroll
        for (int m = 0; m < 4; ++m) {
            aNxt[m].h[0] = *(const v8bf*)(aRow[m] + k0 + o0);
            aNxt[m].h[1] = *(const v8bf*)(aRow[m] + k0 + o1);
        }
#pragma unroll
        for (int m = 0; m < 4; ++m)
            acc[m] = __builtin_amdgcn_wmma_f32_16x16x32_bf16(
                false, aCur[m].v, false, bCur.v, (short)0, acc[m], false, false);
        bCur = bNxt;
#pragma unroll
        for (int m = 0; m < 4; ++m) aCur[m] = aNxt[m];
    }

    // epilogue: final K-step
#pragma unroll
    for (int m = 0; m < 4; ++m)
        acc[m] = __builtin_amdgcn_wmma_f32_16x16x32_bf16(
            false, aCur[m].v, false, bCur.v, (short)0, acc[m], false, false);

    const int col = nBase + r;
    const float bv = bias ? bias[col] : 0.0f;
    float* cc = C + (size_t)col;
#pragma unroll
    for (int m = 0; m < 4; ++m) {
#pragma unroll
        for (int v = 0; v < 8; ++v) {
            int row = mBase + 16 * m + v + half * 8;
            cc[(size_t)row * (size_t)ldc] = acc[m][v] + bv;
        }
    }
}

// ---------------- attention query prep: q[b,:] = [h0+h1, h2+h3] (bf16) ----------------
__global__ __launch_bounds__(256) void k_qprep(const float* __restrict__ hidden,
                                               bf16_t* __restrict__ q) {
    int i = blockIdx.x * blockDim.x + threadIdx.x;   // B*H2
    if (i >= BB * H2C) return;
    int b = i >> 10, j = i & (H2C - 1);
    int grp = j >> 9, h = j & (HH - 1);
    float v = hidden[((size_t)(grp * 2 + 0) * BB + b) * HH + h]
            + hidden[((size_t)(grp * 2 + 1) * BB + b) * HH + h];
    q[(size_t)b * H2C + j] = (bf16_t)v;
}

// ---------------- scores[b,s] = Va . tanh(qw[b]+keyed[b,s]) + bV, masked ----------------
__global__ __launch_bounds__(128) void k_scores(const float* __restrict__ qw,
                                                const float* __restrict__ keyed,
                                                const float* __restrict__ Va,
                                                const float* __restrict__ bV,
                                                const int* __restrict__ mask,
                                                float* __restrict__ scores) {
    int widx = blockIdx.x * (blockDim.x >> 5) + (threadIdx.x >> 5);
    int lane = threadIdx.x & 31;
    if (widx >= BB * SS) return;
    int b = widx / SS, s = widx % SS;
    const float* qr = qw + (size_t)b * H2C;
    const float* kr = keyed + ((size_t)b * SS + s) * H2C;
    float acc = 0.0f;
    for (int e = lane; e < H2C; e += 32)
        acc += Va[e] * tanhf(qr[e] + kr[e]);
    for (int off = 16; off; off >>= 1) acc += __shfl_down(acc, off, 32);
    if (lane == 0) {
        float sc = acc + bV[0];
        if (mask[b * SS + s] == 0) sc = -1e10f;
        scores[b * SS + s] = sc;
    }
}

// ---------------- softmax over S + context = alpha . enc ----------------
__global__ __launch_bounds__(128) void k_softmax_ctx(const float* __restrict__ scores,
                                                     const float* __restrict__ enc,
                                                     float* __restrict__ context) {
    int b = blockIdx.x, t = threadIdx.x;
    __shared__ float sv[SS];
    __shared__ float al[SS];
    float x = scores[b * SS + t];
    sv[t] = x; __syncthreads();
    for (int off = 64; off; off >>= 1) { if (t < off) sv[t] = fmaxf(sv[t], sv[t + off]); __syncthreads(); }
    float m = sv[0]; __syncthreads();
    float e = __expf(x - m);
    sv[t] = e; __syncthreads();
    for (int off = 64; off; off >>= 1) { if (t < off) sv[t] += sv[t + off]; __syncthreads(); }
    float inv = 1.0f / sv[0];
    al[t] = e * inv; __syncthreads();
    const float* eb = enc + (size_t)b * SS * H2C;
    for (int d = t; d < H2C; d += 128) {
        float acc = 0.0f;
        for (int s = 0; s < SS; ++s) acc += al[s] * eb[(size_t)s * H2C + d];
        context[(size_t)b * H2C + d] = acc;
    }
}

// ---------------- embedded = leaky_relu(emb[tok]@linW^T + linb); x = [embedded, context] ----------------
__global__ __launch_bounds__(256) void k_embed_x(const long long* __restrict__ target,
                                                 const int* __restrict__ sos, int t,
                                                 const float* __restrict__ emb,
                                                 const float* __restrict__ linW,
                                                 const float* __restrict__ linb,
                                                 const float* __restrict__ context,
                                                 bf16_t* __restrict__ x) {
    int i = blockIdx.x * blockDim.x + threadIdx.x;   // B*H2
    if (i >= BB * H2C) return;
    int b = i >> 10, j = i & (H2C - 1);
    long long tok = (t == 0) ? (long long)sos[0] : target[(size_t)b * TT + (t - 1)];
    const float* ev = emb + (size_t)tok * EE;
    const float* wr = linW + (size_t)j * EE;
    float acc = linb[j];
    for (int e = 0; e < EE; ++e) acc += ev[e] * wr[e];
    float val = acc > 0.0f ? acc : 0.03f * acc;
    x[(size_t)b * H4C + j] = (bf16_t)val;
    x[(size_t)b * H4C + H2C + j] = (bf16_t)context[i];
}

// ---------------- GRU gate fusion (PyTorch r,z,n order) ----------------
__global__ __launch_bounds__(256) void k_gru_gate(const float* __restrict__ gi,
                                                  const float* __restrict__ gh,
                                                  float* __restrict__ hidden,
                                                  bf16_t* __restrict__ hidden_bf,
                                                  bf16_t* __restrict__ y, int layer) {
    int i = blockIdx.x * blockDim.x + threadIdx.x;   // 2*B*H
    if (i >= 2 * BB * HH) return;
    int d = i >> 15;
    int rem = i & (BB * HH - 1);
    int b = rem >> 9, h = rem & (HH - 1);
    int slot = layer * 2 + d;
    const float* gib = gi + (size_t)b * N3 + (size_t)d * G3;
    const float* ghb = gh + (size_t)b * N3 + (size_t)d * G3;
    float rg = 1.0f / (1.0f + __expf(-(gib[h] + ghb[h])));
    float zg = 1.0f / (1.0f + __expf(-(gib[HH + h] + ghb[HH + h])));
    float ng = tanhf(gib[2 * HH + h] + rg * ghb[2 * HH + h]);
    size_t hoff = ((size_t)slot * BB + b) * HH + h;
    float hold = hidden[hoff];
    float hnew = (1.0f - zg) * ng + zg * hold;
    hidden[hoff] = hnew;
    hidden_bf[hoff] = (bf16_t)hnew;
    y[(size_t)b * H2C + (size_t)d * HH + h] = (bf16_t)hnew;
}

// ---------------- in-place log_softmax over V for one step ----------------
__global__ __launch_bounds__(256) void k_logsoftmax(float* __restrict__ out, int t) {
    int b = blockIdx.x, tid = threadIdx.x;
    float* row = out + ((size_t)b * TT + t) * VV;
    __shared__ float red[256];
    float m = -1e30f;
    for (int v = tid; v < VV; v += 256) m = fmaxf(m, row[v]);
    red[tid] = m; __syncthreads();
    for (int off = 128; off; off >>= 1) { if (tid < off) red[tid] = fmaxf(red[tid], red[tid + off]); __syncthreads(); }
    m = red[0]; __syncthreads();
    float s = 0.0f;
    for (int v = tid; v < VV; v += 256) s += __expf(row[v] - m);
    red[tid] = s; __syncthreads();
    for (int off = 128; off; off >>= 1) { if (tid < off) red[tid] += red[tid + off]; __syncthreads(); }
    float lse = m + logf(red[0]);
    for (int v = tid; v < VV; v += 256) row[v] -= lse;
}

// ---------------- host orchestration ----------------
extern "C" void kernel_launch(void* const* d_in, const int* in_sizes, int n_in,
                              void* d_out, int out_size, void* d_ws, size_t ws_size,
                              hipStream_t stream) {
    const float*     enc  = (const float*)d_in[0];
    const float*     ehid = (const float*)d_in[1];
    const int*       mask = (const int*)d_in[2];
    const long long* tgt  = (const long long*)d_in[3];
    const int*       sos  = (const int*)d_in[4];
    const float*     emb  = (const float*)d_in[6];
    const float*     linW = (const float*)d_in[7];
    const float*     linb = (const float*)d_in[8];
    const float*     W1   = (const float*)d_in[9];
    const float*     b1   = (const float*)d_in[10];
    const float*     W2   = (const float*)d_in[11];
    const float*     b2   = (const float*)d_in[12];
    const float*     Va   = (const float*)d_in[13];
    const float*     bV   = (const float*)d_in[14];
    const float*     Wih0 = (const float*)d_in[15];
    const float*     Whh0 = (const float*)d_in[16];
    const float*     bih0 = (const float*)d_in[17];
    const float*     bhh0 = (const float*)d_in[18];
    const float*     Wih1 = (const float*)d_in[19];
    const float*     Whh1 = (const float*)d_in[20];
    const float*     bih1 = (const float*)d_in[21];
    const float*     bhh1 = (const float*)d_in[22];
    const float*     outW = (const float*)d_in[23];
    const float*     outb = (const float*)d_in[24];
    float* out = (float*)d_out;

    // bump allocator over workspace
    char* p = (char*)d_ws;
    auto alloc = [&](size_t bytes) -> char* {
        char* r = p; p += (bytes + 255) & ~(size_t)255; return r;
    };

    bf16_t* enc_bf   = (bf16_t*)alloc((size_t)BB * SS * H2C * 2);
    bf16_t* W1_bf    = (bf16_t*)alloc((size_t)H2C * H2C * 2);
    bf16_t* W2_bf    = (bf16_t*)alloc((size_t)H2C * H2C * 2);
    bf16_t* Wih0_bf  = (bf16_t*)alloc((size_t)N3 * H4C * 2);
    bf16_t* Whh0_bf  = (bf16_t*)alloc((size_t)N3 * HH * 2);
    bf16_t* Wih1_bf  = (bf16_t*)alloc((size_t)N3 * H2C * 2);
    bf16_t* Whh1_bf  = (bf16_t*)alloc((size_t)N3 * HH * 2);
    bf16_t* outW_bf  = (bf16_t*)alloc((size_t)VV * H2C * 2);
    float*  keyed    = (float*)alloc((size_t)BB * SS * H2C * 4);
    float*  hid_f    = (float*)alloc((size_t)4 * BB * HH * 4);
    bf16_t* hid_bf   = (bf16_t*)alloc((size_t)4 * BB * HH * 2);
    bf16_t* q_bf     = (bf16_t*)alloc((size_t)BB * H2C * 2);
    float*  qw       = (float*)alloc((size_t)BB * H2C * 4);
    float*  scores   = (float*)alloc((size_t)BB * SS * 4);
    float*  context  = (float*)alloc((size_t)BB * H2C * 4);
    bf16_t* x_bf     = (bf16_t*)alloc((size_t)BB * H4C * 2);
    float*  gi0      = (float*)alloc((size_t)BB * N3 * 4);
    float*  gh0      = (float*)alloc((size_t)BB * N3 * 4);
    bf16_t* y0_bf    = (bf16_t*)alloc((size_t)BB * H2C * 2);
    float*  gi1      = (float*)alloc((size_t)BB * N3 * 4);
    float*  gh1      = (float*)alloc((size_t)BB * N3 * 4);
    bf16_t* y1_bf    = (bf16_t*)alloc((size_t)BB * H2C * 2);

    auto cvt = [&](const float* s, bf16_t* d, size_t n) {
        k_cvt<<<(unsigned)((n + 255) / 256), 256, 0, stream>>>(s, d, n);
    };
    auto gemm = [&](const bf16_t* A, const bf16_t* Bt, const float* bias, float* C,
                    int M, int N, int K, long long ldc) {
        dim3 g((unsigned)(N / 64), (unsigned)(M / 64));
        wmma_gemm_bf16<<<g, 128, 0, stream>>>(A, Bt, bias, C, M, N, K, ldc);
    };

    // ---- one-time precompute ----
    cvt(enc,  enc_bf,  (size_t)BB * SS * H2C);
    cvt(W1,   W1_bf,   (size_t)H2C * H2C);
    cvt(W2,   W2_bf,   (size_t)H2C * H2C);
    cvt(Wih0, Wih0_bf, (size_t)N3 * H4C);
    cvt(Whh0, Whh0_bf, (size_t)N3 * HH);
    cvt(Wih1, Wih1_bf, (size_t)N3 * H2C);
    cvt(Whh1, Whh1_bf, (size_t)N3 * HH);
    cvt(outW, outW_bf, (size_t)VV * H2C);

    // keyed = enc @ W2^T + b2   (M = B*S)
    gemm(enc_bf, W2_bf, b2, keyed, BB * SS, H2C, H2C, H2C);

    // init hidden (fp32 + bf16)
    k_copy<<<(4 * BB * HH + 255) / 256, 256, 0, stream>>>(ehid, hid_f, 4 * BB * HH);
    cvt(hid_f, hid_bf, (size_t)4 * BB * HH);

    const int EW = (BB * H2C + 255) / 256;   // elementwise grid (65536 threads)

    for (int t = 0; t < TT; ++t) {
        // attention
        k_qprep<<<EW, 256, 0, stream>>>(hid_f, q_bf);
        gemm(q_bf, W1_bf, b1, qw, BB, H2C, H2C, H2C);
        k_scores<<<(BB * SS) / 4, 128, 0, stream>>>(qw, keyed, Va, bV, mask, scores);
        k_softmax_ctx<<<BB, 128, 0, stream>>>(scores, enc, context);

        // embedding + input assembly
        k_embed_x<<<EW, 256, 0, stream>>>(tgt, sos, t, emb, linW, linb, context, x_bf);

        // GRU layer 0
        gemm(x_bf, Wih0_bf, bih0, gi0, BB, N3, H4C, N3);
        for (int d = 0; d < 2; ++d)
            gemm(hid_bf + (size_t)d * BB * HH, Whh0_bf + (size_t)d * G3 * HH,
                 bhh0 + (size_t)d * G3, gh0 + (size_t)d * G3, BB, G3, HH, N3);
        k_gru_gate<<<EW, 256, 0, stream>>>(gi0, gh0, hid_f, hid_bf, y0_bf, 0);

        // GRU layer 1
        gemm(y0_bf, Wih1_bf, bih1, gi1, BB, N3, H2C, N3);
        for (int d = 0; d < 2; ++d)
            gemm(hid_bf + (size_t)(2 + d) * BB * HH, Whh1_bf + (size_t)d * G3 * HH,
                 bhh1 + (size_t)d * G3, gh1 + (size_t)d * G3, BB, G3, HH, N3);
        k_gru_gate<<<EW, 256, 0, stream>>>(gi1, gh1, hid_f, hid_bf, y1_bf, 1);

        // logits directly into d_out (strided rows), then in-place log_softmax
        gemm(y1_bf, outW_bf, outb, out + (size_t)t * VV, BB, VV, H2C, (long long)TT * VV);
        k_logsoftmax<<<BB, 256, 0, stream>>>(out, t);
    }

    // hidden_final appended after log_probs
    k_copy<<<(4 * BB * HH + 255) / 256, 256, 0, stream>>>(
        hid_f, out + (size_t)BB * TT * VV, 4 * BB * HH);
}
